// Block_48936857370743
// MI455X (gfx1250) — matrix operand order repaired and checked by reading, not compiled, not run on hardware
//
#include <hip/hip_runtime.h>
#include <hip/hip_bf16.h>

#ifndef __has_builtin
#define __has_builtin(x) 0
#endif

// ---------------------------------------------------------------------------
// Types for CDNA5 WMMA (gfx1250, wave32)
// ---------------------------------------------------------------------------
typedef __attribute__((ext_vector_type(16))) __bf16 bf16x16;
typedef __attribute__((ext_vector_type(8)))  __bf16 bf16x8;
typedef __attribute__((ext_vector_type(8)))  float  f32x8;
typedef int v4i __attribute__((vector_size(16)));

static __device__ __forceinline__ f32x8 wmma_bf16(bf16x16 a, bf16x16 b, f32x8 c) {
    // D = A(16x32) * B(32x16) + C(16x16 f32)
    return __builtin_amdgcn_wmma_f32_16x16x32_bf16(
        /*neg_a=*/false, a, /*neg_b=*/false, b,
        /*c_mod=*/(short)0, c, /*reuse_a=*/false, /*reuse_b=*/false);
}

// bf16 <-> f32 via bit ops (round-to-nearest-even), independent of native casts
static __device__ __forceinline__ __bf16 f2bf(float f) {
    unsigned u = __float_as_uint(f);
    unsigned r = u + 0x7fffu + ((u >> 16) & 1u);
    unsigned short h = (unsigned short)(r >> 16);
    union { unsigned short s; __bf16 b; } cv; cv.s = h; return cv.b;
}
static __device__ __forceinline__ __bf16 to_bf(float v) { return f2bf(v); }
static __device__ __forceinline__ __bf16 to_bf(__bf16 v) { return v; }

// ---------------------------------------------------------------------------
// CDNA5 async global->LDS copy (GLOBAL_LOAD_ASYNC_TO_LDS_B128, ASYNCcnt).
// Builtin signature (probe-confirmed by round-2 diagnostic):
//   (v4i __attribute__((address_space(1)))* src, v4i AS(3)* dst, imm, imm)
// Falls back to a synchronous reg-staged copy if unavailable.
// ---------------------------------------------------------------------------
#if __has_builtin(__builtin_amdgcn_global_load_async_to_lds_b128)
#define HAVE_ASYNC_LDS 1
#else
#define HAVE_ASYNC_LDS 0
#endif

static __device__ __forceinline__ void async_cp16(void* lds_dst, const void* gsrc) {
#if HAVE_ASYNC_LDS
    __builtin_amdgcn_global_load_async_to_lds_b128(
        (__attribute__((address_space(1))) v4i*)gsrc,
        (__attribute__((address_space(3))) v4i*)lds_dst, 0, 0);
#else
    *(bf16x8*)lds_dst = *(const bf16x8*)gsrc;
#endif
}
static __device__ __forceinline__ void async_wait0() {
#if HAVE_ASYNC_LDS
    asm volatile("s_wait_asynccnt 0x0" ::: "memory");
#endif
}

// ---------------------------------------------------------------------------
// Fragment loads (wave32 layouts per CDNA5 ISA 7.12.2)
// A 16x32 bf16: lane<16: row=lane, K in {0..7, 16..23}; lane>=16: row=lane-16,
//               K in {8..15, 24..31}. Two contiguous 8-elt (16B) chunks.
// B 32x16 bf16 (as Bt rows, N x K): lane n=lane&15, K same as A:
//               lane<16: {0..7, 16..23}; lane>=16: {8..15, 24..31}.
// ---------------------------------------------------------------------------
static __device__ __forceinline__ bf16x16 load_frag_a(const __bf16* base, int ld) {
    int lane = threadIdx.x & 31;
    int row  = lane & 15;
    int koff = (lane < 16) ? 0 : 8;
    const __bf16* p = base + (size_t)row * ld + koff;
    bf16x8 lo = *(const bf16x8*)(p);
    bf16x8 hi = *(const bf16x8*)(p + 16);
    return __builtin_shufflevector(lo, hi, 0,1,2,3,4,5,6,7,8,9,10,11,12,13,14,15);
}
static __device__ __forceinline__ bf16x16 load_frag_b(const __bf16* base, int ld) {
    int lane = threadIdx.x & 31;
    int n    = lane & 15;
    int koff = (lane < 16) ? 0 : 8;
    const __bf16* p = base + (size_t)n * ld + koff;
    bf16x8 lo = *(const bf16x8*)(p);
    bf16x8 hi = *(const bf16x8*)(p + 16);
    return __builtin_shufflevector(lo, hi, 0,1,2,3,4,5,6,7,8,9,10,11,12,13,14,15);
}

// ---------------------------------------------------------------------------
// Generic batched bf16 GEMM:  C[M,N] (+Res) = A[M,K] @ Bt[N,K]^T
// Block tile 128xBN (BN=128 or 64), 8 waves (4x2), wave tile 32x(BN/2).
// Double-buffered LDS staging via async global->LDS, one barrier per K-step.
// ---------------------------------------------------------------------------
template<int BN, int OUT_BF16, int RELU, int ADD_RES>
__global__ __launch_bounds__(256) void gemm_bt(
    const __bf16* __restrict__ A, const __bf16* __restrict__ Bt,
    void* __restrict__ Cout, const float* __restrict__ Res,
    int K, int lda, int ldb, int ldc,
    int b2, long long sa1, long long sa2, long long sb1, long long sb2,
    long long sc1, long long sc2)
{
    constexpr int WN = BN / 2;      // wave N tile
    constexpr int NF = WN / 16;     // B fragments per wave (4 or 2)

    int bz = blockIdx.z;
    int i1 = bz / b2, i2 = bz - i1 * b2;
    A  += (size_t)i1 * sa1 + (size_t)i2 * sa2;
    Bt += (size_t)i1 * sb1 + (size_t)i2 * sb2;
    size_t cbase = (size_t)i1 * sc1 + (size_t)i2 * sc2;

    int m0 = blockIdx.y * 128;
    int n0 = blockIdx.x * BN;

    __shared__ __attribute__((aligned(16))) __bf16 As[2][128 * 32];
    __shared__ __attribute__((aligned(16))) __bf16 Bs[2][BN * 32];

    int wave = threadIdx.x >> 5;
    int wr = wave >> 1;          // 0..3  -> M sub-tile (32 rows)
    int wc = wave & 1;           // 0..1  -> N sub-tile (WN cols)
    int lane = threadIdx.x & 31;
    int half8 = (lane >> 4) << 3;
    int cl = lane & 15;

    // staging thread mapping
    int ar = threadIdx.x >> 1, ac = (threadIdx.x & 1) * 16;           // A: 128x32
    int br, bc;
    if (BN == 128) { br = threadIdx.x >> 1; bc = (threadIdx.x & 1) * 16; }
    else           { br = threadIdx.x >> 2; bc = (threadIdx.x & 3) * 8; }

    auto stage = [&](int buf, int k0) {
        const __bf16* asrc = A + (size_t)(m0 + ar) * lda + k0 + ac;
        async_cp16(&As[buf][ar * 32 + ac],     asrc);
        async_cp16(&As[buf][ar * 32 + ac + 8], asrc + 8);
        const __bf16* bsrc = Bt + (size_t)(n0 + br) * ldb + k0 + bc;
        if (BN == 128) {
            async_cp16(&Bs[buf][br * 32 + bc],     bsrc);
            async_cp16(&Bs[buf][br * 32 + bc + 8], bsrc + 8);
        } else {
            async_cp16(&Bs[buf][br * 32 + bc], bsrc);
        }
    };

    f32x8 acc[2][NF];
    for (int mi = 0; mi < 2; mi++)
        for (int nf = 0; nf < NF; nf++)
            for (int j = 0; j < 8; j++) acc[mi][nf][j] = 0.f;

    int nk = K >> 5;
    stage(0, 0);
    async_wait0();
    __syncthreads();

    for (int i = 0; i < nk; i++) {
        int cur = i & 1, nxt = cur ^ 1;
        if (i + 1 < nk) stage(nxt, (i + 1) * 32);

        bf16x16 a0 = load_frag_a(&As[cur][(wr * 32)      * 32], 32);
        bf16x16 a1 = load_frag_a(&As[cur][(wr * 32 + 16) * 32], 32);
        #pragma unroll
        for (int nf = 0; nf < NF; nf++) {
            bf16x16 bfr = load_frag_b(&Bs[cur][(wc * WN + nf * 16) * 32], 32);
            acc[0][nf] = wmma_bf16(a0, bfr, acc[0][nf]);
            acc[1][nf] = wmma_bf16(a1, bfr, acc[1][nf]);
        }
        async_wait0();
        __syncthreads();
    }

    #pragma unroll
    for (int mi = 0; mi < 2; mi++)
        #pragma unroll
        for (int nf = 0; nf < NF; nf++)
            #pragma unroll
            for (int j = 0; j < 8; j++) {
                int row = m0 + wr * 32 + mi * 16 + j + half8;
                int col = n0 + wc * WN + nf * 16 + cl;
                size_t idx = cbase + (size_t)row * ldc + col;
                float v = acc[mi][nf][j];
                if (ADD_RES) v += Res[idx];
                if (RELU)    v = fmaxf(v, 0.f);
                if (OUT_BF16) ((__bf16*)Cout)[idx] = f2bf(v);
                else          ((float*)Cout)[idx]  = v;
            }
}

// ---------------------------------------------------------------------------
// Fused causal MLA attention over latent z:
//   att = q_lat @ z^T / sqrt(Dh);  P = softmax(causal(att));  S = P @ z
// grid (T/64, B*H), 4 waves/block, each wave owns 16 q rows (flash style).
// ---------------------------------------------------------------------------
__global__ __launch_bounds__(128) void mla_attn_kernel(
    const __bf16* __restrict__ qlat, const __bf16* __restrict__ z,
    const __bf16* __restrict__ zT, __bf16* __restrict__ S, int T)
{
    const int L = 64, H = 16;
    int bh = blockIdx.y;
    int b  = bh / H;
    const __bf16* qb  = qlat + (size_t)bh * T * L;
    __bf16*       Sb  = S    + (size_t)bh * T * L;
    const __bf16* zb  = z    + (size_t)b  * T * L;
    const __bf16* ztb = zT   + (size_t)b  * L * T;

    int wave = threadIdx.x >> 5;
    int lane = threadIdx.x & 31;
    int half8 = (lane >> 4) << 3;
    int cl = lane & 15;
    int t0 = blockIdx.x * 64 + wave * 16;

    __shared__ __attribute__((aligned(16))) __bf16 plds[4][16 * 32];
    __bf16* myp = plds[wave];

    bf16x16 aq0 = load_frag_a(qb + (size_t)t0 * L, L);
    bf16x16 aq1 = load_frag_a(qb + (size_t)t0 * L + 32, L);

    f32x8 o[4];
    for (int f = 0; f < 4; f++) for (int j = 0; j < 8; j++) o[f][j] = 0.f;
    float mrow[8], lrow[8];
    for (int j = 0; j < 8; j++) { mrow[j] = -3.0e38f; lrow[j] = 0.f; }

    const float scale = 0.08838834764831845f;   // 1/sqrt(128)
    int smax = t0 + 15;
    for (int s0 = 0; s0 <= smax; s0 += 32) {
        f32x8 att[2];
        #pragma unroll
        for (int nt = 0; nt < 2; nt++) {
            bf16x16 b0 = load_frag_b(zb + (size_t)(s0 + nt * 16) * L, L);
            bf16x16 b1 = load_frag_b(zb + (size_t)(s0 + nt * 16) * L + 32, L);
            f32x8 c; for (int j = 0; j < 8; j++) c[j] = 0.f;
            c = wmma_bf16(aq0, b0, c);
            c = wmma_bf16(aq1, b1, c);
            att[nt] = c;
        }
        #pragma unroll
        for (int j = 0; j < 8; j++) {
            int rg = t0 + j + half8;
            float v0 = att[0][j] * scale; if (s0 + cl      > rg) v0 = -3.0e38f;
            float v1 = att[1][j] * scale; if (s0 + 16 + cl > rg) v1 = -3.0e38f;
            float mx = fmaxf(v0, v1);
            #pragma unroll
            for (int off = 8; off > 0; off >>= 1) mx = fmaxf(mx, __shfl_xor(mx, off, 32));
            float nm   = fmaxf(mrow[j], mx);
            float corr = __expf(mrow[j] - nm);
            float p0 = __expf(v0 - nm);
            float p1 = __expf(v1 - nm);
            float rs = p0 + p1;
            #pragma unroll
            for (int off = 8; off > 0; off >>= 1) rs += __shfl_xor(rs, off, 32);
            lrow[j] = lrow[j] * corr + rs;
            mrow[j] = nm;
            o[0][j] *= corr; o[1][j] *= corr; o[2][j] *= corr; o[3][j] *= corr;
            int rl = j + half8;                 // restage P (C-layout -> A-layout)
            myp[rl * 32 + cl]      = f2bf(p0);
            myp[rl * 32 + 16 + cl] = f2bf(p1);
        }
        asm volatile("s_wait_dscnt 0x0" ::: "memory");
        bf16x16 pa = load_frag_a(myp, 32);
        #pragma unroll
        for (int lc = 0; lc < 4; lc++) {
            bf16x16 bz = load_frag_b(ztb + (size_t)(lc * 16) * T + s0, T);
            o[lc] = wmma_bf16(pa, bz, o[lc]);
        }
    }
    #pragma unroll
    for (int j = 0; j < 8; j++) {
        float inv = 1.0f / lrow[j];
        int rg = t0 + j + half8;
        #pragma unroll
        for (int lc = 0; lc < 4; lc++)
            Sb[(size_t)rg * L + lc * 16 + cl] = f2bf(o[lc][j] * inv);
    }
}

// ---------------------------------------------------------------------------
// Helpers: rmsnorm (fp32 in -> bf16 out), transpose-to-bf16, convert-to-bf16
// ---------------------------------------------------------------------------
__global__ __launch_bounds__(256) void rmsnorm_kernel(
    const float* __restrict__ x, const float* __restrict__ w,
    __bf16* __restrict__ out, int Cdim)
{
    int row = blockIdx.x;
    const float* xr = x + (size_t)row * Cdim;
    float ss = 0.f;
    for (int i = threadIdx.x; i < Cdim; i += blockDim.x) { float v = xr[i]; ss += v * v; }
    __shared__ float red[256];
    red[threadIdx.x] = ss;
    __syncthreads();
    for (int s = 128; s > 0; s >>= 1) {
        if (threadIdx.x < (unsigned)s) red[threadIdx.x] += red[threadIdx.x + s];
        __syncthreads();
    }
    float inv = rsqrtf(red[0] / (float)Cdim + 5e-5f);
    __bf16* orow = out + (size_t)row * Cdim;
    for (int i = threadIdx.x; i < Cdim; i += blockDim.x)
        orow[i] = f2bf(xr[i] * inv * w[i]);
}

template<typename TIN>
__global__ __launch_bounds__(256) void transpose_bf16_kernel(
    const TIN* __restrict__ in, __bf16* __restrict__ out,
    int R, int Cc, long long sin_b, long long sout_b, int nbatch)
{
    long long total = (long long)nbatch * R * Cc;
    long long idx = (long long)blockIdx.x * blockDim.x + threadIdx.x;
    if (idx >= total) return;
    int bidx = (int)(idx / ((long long)R * Cc));
    long long rem = idx - (long long)bidx * R * Cc;
    int r = (int)(rem / Cc);
    int c = (int)(rem - (long long)r * Cc);
    out[(size_t)bidx * sout_b + (size_t)c * R + r] =
        to_bf(in[(size_t)bidx * sin_b + (size_t)r * Cc + c]);
}

__global__ __launch_bounds__(256) void cvt_bf16_kernel(
    const float* __restrict__ in, __bf16* __restrict__ out, long long n)
{
    long long i = (long long)blockIdx.x * blockDim.x + threadIdx.x;
    if (i < n) out[i] = f2bf(in[i]);
}

// ---------------------------------------------------------------------------
// Orchestration
// ---------------------------------------------------------------------------
extern "C" void kernel_launch(void* const* d_in, const int* in_sizes, int n_in,
                              void* d_out, int out_size, void* d_ws, size_t ws_size,
                              hipStream_t stream) {
    (void)in_sizes; (void)n_in; (void)out_size; (void)ws_size;
    const int B = 2, T = 2048, C = 2048, H = 16, Dh = 128, L = 64, F = 8192;
    const int M = B * T;

    const float*  x     = (const float*)d_in[0];
    const float*  wq    = (const float*)d_in[1];
    const float*  wkv   = (const float*)d_in[2];
    const float*  wkup  = (const float*)d_in[3];
    const float*  wvup  = (const float*)d_in[4];
    const float*  wo    = (const float*)d_in[5];
    const float*  ln1   = (const float*)d_in[6];
    const float*  ln2   = (const float*)d_in[7];
    const __bf16* wfc   = (const __bf16*)d_in[8];
    const __bf16* wproj = (const __bf16*)d_in[9];
    float* out = (float*)d_out;

    char* ws = (char*)d_ws;
    size_t off = 0;
    auto alloc_bf = [&](size_t elems) -> __bf16* {
        __bf16* p = (__bf16*)(ws + off);
        off += ((elems * 2 + 255) & ~(size_t)255);
        return p;
    };
    __bf16* h1     = alloc_bf((size_t)M * C);      // h / h2 (reused)
    __bf16* wqT    = alloc_bf((size_t)C * C);
    __bf16* wkvT   = alloc_bf((size_t)L * C);
    __bf16* wkupB  = alloc_bf((size_t)H * L * Dh); // native layout already Bt
    __bf16* wvupT  = alloc_bf((size_t)H * Dh * L);
    __bf16* woT    = alloc_bf((size_t)C * C);
    __bf16* wfcT   = alloc_bf((size_t)F * C);
    __bf16* wprojT = alloc_bf((size_t)C * F);
    __bf16* qb     = alloc_bf((size_t)M * C);
    __bf16* zb     = alloc_bf((size_t)M * L);
    __bf16* zTb    = alloc_bf((size_t)B * L * T);
    __bf16* qlat   = alloc_bf((size_t)B * H * T * L);
    __bf16* Sb     = alloc_bf((size_t)B * H * T * L);
    __bf16* Yb     = alloc_bf((size_t)M * C);
    __bf16* g1     = alloc_bf((size_t)M * F);

    auto nb = [](long long tot) { return (unsigned)((tot + 255) / 256); };

    // --- weight prep (bf16, Bt layout) ---
    transpose_bf16_kernel<float><<<nb((long long)C * C), 256, 0, stream>>>(wq, wqT, C, C, 0, 0, 1);
    transpose_bf16_kernel<float><<<nb((long long)C * L), 256, 0, stream>>>(wkv, wkvT, C, L, 0, 0, 1);
    cvt_bf16_kernel<<<nb((long long)H * L * Dh), 256, 0, stream>>>(wkup, wkupB, (long long)H * L * Dh);
    transpose_bf16_kernel<float><<<nb((long long)H * L * Dh), 256, 0, stream>>>(
        wvup, wvupT, L, Dh, (long long)L * Dh, (long long)Dh * L, H);
    transpose_bf16_kernel<float><<<nb((long long)C * C), 256, 0, stream>>>(wo, woT, C, C, 0, 0, 1);
    transpose_bf16_kernel<__bf16><<<nb((long long)C * F), 256, 0, stream>>>(wfc, wfcT, C, F, 0, 0, 1);
    transpose_bf16_kernel<__bf16><<<nb((long long)F * C), 256, 0, stream>>>(wproj, wprojT, F, C, 0, 0, 1);

    // --- attention sub-block ---
    rmsnorm_kernel<<<M, 256, 0, stream>>>(x, ln1, h1, C);
    gemm_bt<128,1,0,0><<<dim3(C / 128, M / 128, 1), 256, 0, stream>>>(
        h1, wqT, qb, (const float*)nullptr, C, C, C, C, 1, 0, 0, 0, 0, 0, 0);
    gemm_bt<64,1,0,0><<<dim3(1, M / 128, 1), 256, 0, stream>>>(
        h1, wkvT, zb, (const float*)nullptr, C, C, C, L, 1, 0, 0, 0, 0, 0, 0);
    transpose_bf16_kernel<__bf16><<<nb((long long)B * T * L), 256, 0, stream>>>(
        zb, zTb, T, L, (long long)T * L, (long long)L * T, B);
    // q_lat[b,h] = q[b,:,h,:] @ Wk_up[h]^T   (Wk_up native = Bt)
    gemm_bt<64,1,0,0><<<dim3(1, T / 128, B * H), 256, 0, stream>>>(
        qb, wkupB, qlat, (const float*)nullptr, Dh, C, Dh, L, H,
        (long long)T * C, (long long)Dh, 0, (long long)L * Dh,
        (long long)H * T * L, (long long)T * L);
    mla_attn_kernel<<<dim3(T / 64, B * H), 128, 0, stream>>>(qlat, zb, zTb, Sb, T);
    // Y[b,:,h,:] = S[b,h] @ Wv_up[h]
    gemm_bt<128,1,0,0><<<dim3(1, T / 128, B * H), 256, 0, stream>>>(
        Sb, wvupT, Yb, (const float*)nullptr, L, L, L, C, H,
        (long long)H * T * L, (long long)T * L, 0, (long long)Dh * L,
        (long long)T * C, (long long)Dh);
    // x1 = x + Y @ wo   (fp32 out -> d_out)
    gemm_bt<128,0,0,1><<<dim3(C / 128, M / 128, 1), 256, 0, stream>>>(
        Yb, woT, out, x, C, C, C, C, 1, 0, 0, 0, 0, 0, 0);

    // --- MLP sub-block ---
    rmsnorm_kernel<<<M, 256, 0, stream>>>(out, ln2, h1, C);
    gemm_bt<128,1,1,0><<<dim3(F / 128, M / 128, 1), 256, 0, stream>>>(
        h1, wfcT, g1, (const float*)nullptr, C, C, C, F, 1, 0, 0, 0, 0, 0, 0);
    gemm_bt<128,0,0,1><<<dim3(C / 128, M / 128, 1), 256, 0, stream>>>(
        g1, wprojT, out, out, F, F, F, C, 1, 0, 0, 0, 0, 0, 0);
}
